// HierarchicalMambaStage_3547642986688
// MI455X (gfx1250) — matrix operand order repaired and checked by reading, not compiled, not run on hardware
//
#include <hip/hip_runtime.h>
#include <hip/hip_bf16.h>

// HierarchicalMambaStage for MI455X (gfx1250, wave32).
// All dense GEMMs run through v_wmma_f32_16x16x32_f16 (f16 in / f32 acc).
// Round 2: 32x64 wave tile + register double-buffered K pipeline so the
// scheduler can overlap global_load_b128 with v_wmma instead of
// s_wait_loadcnt 0x0 before every WMMA.
// Fragment layouts used here (A/B in contiguous k order, not the ISA table order):
//   A (16x32 f16): lane<16 -> row M=lane, K 0..15; lane>=16 -> K 16..31
//   B (32x16 f16): lane<16 -> col N=lane, K 0..15 (contiguous halves of W row); lane>=16 -> K 16..31
//   C/D (16x16 f32): VGPR r -> M = r + 8*(lane>=16), N = lane&15
// Workspace requirement: ~195 MB.

typedef __attribute__((ext_vector_type(16))) _Float16 v16h;
typedef __attribute__((ext_vector_type(8)))  _Float16 v8h;
typedef __attribute__((ext_vector_type(8)))  float    v8f;

namespace hms {
constexpr int B = 8, L = 1024, D = 512, NB = 4;
constexpr int DIN = 2 * D;           // 1024
constexpr int N = 16, DTR = 32, DCONV = 4, FF = 2 * D;
constexpr int M = B * L;             // 8192 rows for all GEMMs
}

__device__ __forceinline__ v16h hms_load_afrag(const _Float16* __restrict__ ap, int k)
{
    v8h x0 = *(const v8h*)(ap + k);
    v8h x1 = *(const v8h*)(ap + k + 8);
    v16h af;
#pragma unroll
    for (int i = 0; i < 8; ++i) { af[i] = x0[i]; af[i + 8] = x1[i]; }
    return af;
}

// ---------------------------------------------------------------------------
// WMMA GEMM: Y[m,e] = act( sum_k A16[m,k]*W16[e,k] + bias[e] ) (+ resid)
// Wave tile: 32(M) x 64(N) = 8 accumulators; double-buffered K pipeline.
// Block = 128 threads = 4 waves stacked along M (128 rows per block).
// act: 0 none, 1 silu, 2 gelu(exact), 3 softplus.
// ---------------------------------------------------------------------------
__global__ __launch_bounds__(128)
void hms_wmma_gemm(const _Float16* __restrict__ A16, int lda, int pad_div,
                   const _Float16* __restrict__ W16, int K,
                   const float* __restrict__ bias,
                   const float* __restrict__ resid,
                   float* __restrict__ Yf, _Float16* __restrict__ Yh,
                   int ldy, int act)
{
    const int lane = threadIdx.x & 31;
    const int wave = threadIdx.x >> 5;
    const int l15  = lane & 15;
    const int hi   = lane >> 4;                     // lane group 0/1
    const int m0   = blockIdx.x * 128 + wave * 32;  // 32 rows per wave
    const int n0   = blockIdx.y * 64;

    int arow0 = m0 + l15;
    int arow1 = m0 + 16 + l15;
    if (pad_div > 0) {                              // remap into zero-padded input
        arow0 += 2 * (arow0 / pad_div);
        arow1 += 2 * (arow1 / pad_div);
    }
    const int kAlo = hi ? 16 : 0;                   // A chunk start within 32-K block
    const int kB   = hi ? 16 : 0;                   // B K half for this lane group
    const _Float16* __restrict__ ap0 = A16 + (size_t)arow0 * lda + kAlo;
    const _Float16* __restrict__ ap1 = A16 + (size_t)arow1 * lda + kAlo;
    const _Float16* __restrict__ wp0 = W16 + (size_t)(n0 +  0 + l15) * K + kB;
    const _Float16* __restrict__ wp1 = W16 + (size_t)(n0 + 16 + l15) * K + kB;
    const _Float16* __restrict__ wp2 = W16 + (size_t)(n0 + 32 + l15) * K + kB;
    const _Float16* __restrict__ wp3 = W16 + (size_t)(n0 + 48 + l15) * K + kB;

    v8f acc[2][4];
#pragma unroll
    for (int r = 0; r < 2; ++r)
#pragma unroll
        for (int j = 0; j < 4; ++j) acc[r][j] = (v8f){};

    // ---- double-buffered K pipeline (K % 32 == 0) ----
    v16h aX0, aX1, bX0, bX1, bX2, bX3;   // buffer X
    v16h aY0, aY1, bY0, bY1, bY2, bY3;   // buffer Y

    aX0 = hms_load_afrag(ap0, 0);
    aX1 = hms_load_afrag(ap1, 0);
    bX0 = *(const v16h*)(wp0);
    bX1 = *(const v16h*)(wp1);
    bX2 = *(const v16h*)(wp2);
    bX3 = *(const v16h*)(wp3);

    int k0 = 0;
    while (true) {
        // prefetch k0+32 into buffer Y while buffer X feeds the WMMAs
        if (k0 + 32 < K) {
            aY0 = hms_load_afrag(ap0, k0 + 32);
            aY1 = hms_load_afrag(ap1, k0 + 32);
            bY0 = *(const v16h*)(wp0 + k0 + 32);
            bY1 = *(const v16h*)(wp1 + k0 + 32);
            bY2 = *(const v16h*)(wp2 + k0 + 32);
            bY3 = *(const v16h*)(wp3 + k0 + 32);
        }
        acc[0][0] = __builtin_amdgcn_wmma_f32_16x16x32_f16(false, aX0, false, bX0, (short)0, acc[0][0], false, false);
        acc[0][1] = __builtin_amdgcn_wmma_f32_16x16x32_f16(false, aX0, false, bX1, (short)0, acc[0][1], false, false);
        acc[0][2] = __builtin_amdgcn_wmma_f32_16x16x32_f16(false, aX0, false, bX2, (short)0, acc[0][2], false, false);
        acc[0][3] = __builtin_amdgcn_wmma_f32_16x16x32_f16(false, aX0, false, bX3, (short)0, acc[0][3], false, false);
        acc[1][0] = __builtin_amdgcn_wmma_f32_16x16x32_f16(false, aX1, false, bX0, (short)0, acc[1][0], false, false);
        acc[1][1] = __builtin_amdgcn_wmma_f32_16x16x32_f16(false, aX1, false, bX1, (short)0, acc[1][1], false, false);
        acc[1][2] = __builtin_amdgcn_wmma_f32_16x16x32_f16(false, aX1, false, bX2, (short)0, acc[1][2], false, false);
        acc[1][3] = __builtin_amdgcn_wmma_f32_16x16x32_f16(false, aX1, false, bX3, (short)0, acc[1][3], false, false);
        k0 += 32;
        if (k0 >= K) break;

        // prefetch k0+32 into buffer X while buffer Y feeds the WMMAs
        if (k0 + 32 < K) {
            aX0 = hms_load_afrag(ap0, k0 + 32);
            aX1 = hms_load_afrag(ap1, k0 + 32);
            bX0 = *(const v16h*)(wp0 + k0 + 32);
            bX1 = *(const v16h*)(wp1 + k0 + 32);
            bX2 = *(const v16h*)(wp2 + k0 + 32);
            bX3 = *(const v16h*)(wp3 + k0 + 32);
        }
        acc[0][0] = __builtin_amdgcn_wmma_f32_16x16x32_f16(false, aY0, false, bY0, (short)0, acc[0][0], false, false);
        acc[0][1] = __builtin_amdgcn_wmma_f32_16x16x32_f16(false, aY0, false, bY1, (short)0, acc[0][1], false, false);
        acc[0][2] = __builtin_amdgcn_wmma_f32_16x16x32_f16(false, aY0, false, bY2, (short)0, acc[0][2], false, false);
        acc[0][3] = __builtin_amdgcn_wmma_f32_16x16x32_f16(false, aY0, false, bY3, (short)0, acc[0][3], false, false);
        acc[1][0] = __builtin_amdgcn_wmma_f32_16x16x32_f16(false, aY1, false, bY0, (short)0, acc[1][0], false, false);
        acc[1][1] = __builtin_amdgcn_wmma_f32_16x16x32_f16(false, aY1, false, bY1, (short)0, acc[1][1], false, false);
        acc[1][2] = __builtin_amdgcn_wmma_f32_16x16x32_f16(false, aY1, false, bY2, (short)0, acc[1][2], false, false);
        acc[1][3] = __builtin_amdgcn_wmma_f32_16x16x32_f16(false, aY1, false, bY3, (short)0, acc[1][3], false, false);
        k0 += 32;
        if (k0 >= K) break;
    }

    // ---- epilogue: bias + activation + optional residual, f32/f16 stores ----
#pragma unroll
    for (int rb = 0; rb < 2; ++rb) {
        const int mb = m0 + rb * 16 + (hi ? 8 : 0);
#pragma unroll
        for (int j = 0; j < 4; ++j) {
            const int e = n0 + j * 16 + l15;
            const float bv = bias ? bias[e] : 0.0f;
#pragma unroll
            for (int r = 0; r < 8; ++r) {
                float v = acc[rb][j][r] + bv;
                if (act == 1)      v = v / (1.0f + __expf(-v));                    // silu
                else if (act == 2) v = 0.5f * v * (1.0f + erff(v * 0.70710678f));  // gelu
                else if (act == 3) v = (v > 20.0f) ? v : log1pf(__expf(v));        // softplus
                const size_t oi = (size_t)(mb + r) * ldy + e;
                if (resid) v += resid[oi];
                if (Yf) Yf[oi] = v;
                if (Yh) Yh[oi] = (_Float16)v;
            }
        }
    }
}

// ---------------------------------------------------------------------------
// Elementwise / support kernels
// ---------------------------------------------------------------------------
__global__ void hms_f32_to_f16(const float* __restrict__ in, _Float16* __restrict__ out, size_t n)
{
    size_t i = (size_t)blockIdx.x * 256 + threadIdx.x;
    if (i < n) out[i] = (_Float16)in[i];
}

// o[dout*1536 + t*512 + din] = tconv_w[dout, din, t]  (D,D,3) -> (E=D, K=3*D)
__global__ void hms_tconv_w_rearrange(const float* __restrict__ w, _Float16* __restrict__ o)
{
    using namespace hms;
    size_t i = (size_t)blockIdx.x * 256 + threadIdx.x;
    if (i >= (size_t)D * D * 3) return;
    int din  = (int)(i % D);
    int t    = (int)((i / D) % 3);
    int dout = (int)(i / ((size_t)3 * D));
    o[i] = (_Float16)w[(size_t)dout * D * 3 + (size_t)din * 3 + t];
}

// xpad16: per batch, row0 = zeros, rows 1..L = x, row L+1 = zeros (f16)
__global__ void hms_build_xpad(const float* __restrict__ x, _Float16* __restrict__ xp)
{
    using namespace hms;
    size_t i = (size_t)blockIdx.x * 256 + threadIdx.x;
    if (i >= (size_t)B * (L + 2) * D) return;
    int d = (int)(i % D);
    size_t t = i / D;
    int l = (int)(t % (L + 2));
    int b = (int)(t / (L + 2));
    _Float16 v = (_Float16)0.0f;
    if (l >= 1 && l <= L) v = (_Float16)x[((size_t)b * L + (l - 1)) * D + d];
    xp[i] = v;
}

// out16[m*cols + c] = (f16) in[m*ld + c]   (dtr slice of xdb)
__global__ void hms_slice_f16(const float* __restrict__ in, int ld,
                              _Float16* __restrict__ out, int cols, int rows)
{
    size_t i = (size_t)blockIdx.x * 256 + threadIdx.x;
    if (i >= (size_t)rows * cols) return;
    int c = (int)(i % cols);
    int m = (int)(i / cols);
    out[i] = (_Float16)in[(size_t)m * ld + c];
}

// LayerNorm over last dim Dn; one block (128 thr) per row; optional f32/f16 outs.
__global__ __launch_bounds__(128)
void hms_layernorm(const float* __restrict__ in, const float* __restrict__ w,
                   const float* __restrict__ b, float* __restrict__ outf,
                   _Float16* __restrict__ outh, int Dn)
{
    __shared__ float s0s[128], s1s[128];
    const int tid = threadIdx.x;
    const float* x = in + (size_t)blockIdx.x * Dn;
    float s0 = 0.0f, s1 = 0.0f;
    for (int i = tid; i < Dn; i += 128) { float v = x[i]; s0 += v; s1 += v * v; }
    s0s[tid] = s0; s1s[tid] = s1; __syncthreads();
    for (int off = 64; off > 0; off >>= 1) {
        if (tid < off) { s0s[tid] += s0s[tid + off]; s1s[tid] += s1s[tid + off]; }
        __syncthreads();
    }
    const float mu = s0s[0] / Dn;
    const float var = s1s[0] / Dn - mu * mu;
    const float rs = rsqrtf(var + 1e-5f);
    for (int i = tid; i < Dn; i += 128) {
        float v = (x[i] - mu) * rs * w[i] + b[i];
        size_t oi = (size_t)blockIdx.x * Dn + i;
        if (outf) outf[oi] = v;
        if (outh) outh[oi] = (_Float16)v;
    }
}

// depthwise causal conv (k=4, left pad 3) + bias + silu; writes f32 and f16
__global__ void hms_dwconv_silu(const float* __restrict__ u_raw,
                                const float* __restrict__ cw,   // (DIN, DCONV)
                                const float* __restrict__ cb,   // (DIN)
                                float* __restrict__ u, _Float16* __restrict__ u16)
{
    using namespace hms;
    size_t i = (size_t)blockIdx.x * 256 + threadIdx.x;
    if (i >= (size_t)M * DIN) return;
    int d  = (int)(i % DIN);
    size_t ml = i / DIN;
    int l = (int)(ml % L);
    int b = (int)(ml / L);
    float s = cb[d];
#pragma unroll
    for (int t = 0; t < DCONV; ++t) {
        int ls = l - (DCONV - 1) + t;
        if (ls >= 0) s += u_raw[((size_t)b * L + ls) * DIN + d] * cw[(size_t)d * DCONV + t];
    }
    s = s / (1.0f + __expf(-s));
    u[i] = s;
    u16[i] = (_Float16)s;
}

// Selective scan: one thread per (b, d) channel; N=16 state in registers.
__global__ __launch_bounds__(256)
void hms_scan(const float* __restrict__ u, const float* __restrict__ dt,
              const float* __restrict__ xdb,      // (M, 64): [0:32) dtr, [32:48) B, [48:64) C
              const float* __restrict__ A_log,    // (DIN, N) for this block
              const float* __restrict__ Dp,       // (DIN)
              float* __restrict__ y)
{
    using namespace hms;
    int idx = blockIdx.x * 256 + threadIdx.x;
    if (idx >= B * DIN) return;
    int d = idx % DIN, b = idx / DIN;
    float A[N], h[N];
#pragma unroll
    for (int n = 0; n < N; ++n) { A[n] = -__expf(A_log[(size_t)d * N + n]); h[n] = 0.0f; }
    const float Dd = Dp[d];
    for (int l = 0; l < L; ++l) {
        const size_t row = (size_t)b * L + l;
        const float ut  = u[row * DIN + d];
        const float dtt = dt[row * DIN + d];
        const float* xr = xdb + row * 64;
        float acc = 0.0f;
#pragma unroll
        for (int n = 0; n < N; ++n) {
            float hb = __expf(dtt * A[n]) * h[n] + dtt * xr[32 + n] * ut;
            h[n] = hb;
            acc += hb * xr[48 + n];
        }
        y[row * DIN + d] = acc + ut * Dd;
    }
}

// a16 = (f16)( y * silu(z) )
__global__ void hms_gate(const float* __restrict__ y, const float* __restrict__ z,
                         _Float16* __restrict__ out, size_t n)
{
    size_t i = (size_t)blockIdx.x * 256 + threadIdx.x;
    if (i >= n) return;
    float zv = z[i];
    out[i] = (_Float16)(y[i] * (zv / (1.0f + __expf(-zv))));
}

// ---------------------------------------------------------------------------
extern "C" void kernel_launch(void* const* d_in, const int* in_sizes, int n_in,
                              void* d_out, int out_size, void* d_ws, size_t ws_size,
                              hipStream_t stream)
{
    using namespace hms;
    (void)in_sizes; (void)n_in; (void)out_size; (void)ws_size;

    const float* x         = (const float*)d_in[0];
    const float* tconv_w   = (const float*)d_in[1];
    const float* tconv_b   = (const float*)d_in[2];
    const float* tnorm_w   = (const float*)d_in[3];
    const float* tnorm_b   = (const float*)d_in[4];
    const float* ln1_w     = (const float*)d_in[5];
    const float* ln1_b     = (const float*)d_in[6];
    const float* in_proj_w = (const float*)d_in[7];
    const float* conv_w    = (const float*)d_in[8];
    const float* conv_b    = (const float*)d_in[9];
    const float* x_proj_w  = (const float*)d_in[10];
    const float* dt_proj_w = (const float*)d_in[11];
    const float* dt_proj_b = (const float*)d_in[12];
    const float* A_log     = (const float*)d_in[13];
    const float* D_param   = (const float*)d_in[14];
    const float* out_proj_w= (const float*)d_in[15];
    const float* ln2_w     = (const float*)d_in[16];
    const float* ln2_b     = (const float*)d_in[17];
    const float* ffn_w1    = (const float*)d_in[18];
    const float* ffn_b1    = (const float*)d_in[19];
    const float* ffn_w2    = (const float*)d_in[20];
    const float* ffn_b2    = (const float*)d_in[21];

    float* hbuf = (float*)d_out;                 // h lives in d_out (B,L,D)

    // --- workspace carve-up (~195 MB) ---
    char* ws = (char*)d_ws;
    size_t off = 0;
    auto alloc_f = [&](size_t n) { float* p = (float*)(ws + off); off = (off + n * 4 + 255) & ~(size_t)255; return p; };
    auto alloc_h = [&](size_t n) { _Float16* p = (_Float16*)(ws + off); off = (off + n * 2 + 255) & ~(size_t)255; return p; };

    float* u_raw = alloc_f((size_t)M * DIN);     // also reused as dt buffer
    float* zbuf  = alloc_f((size_t)M * DIN);     // also holds tconv output at start
    float* ubuf  = alloc_f((size_t)M * DIN);
    float* ybuf  = alloc_f((size_t)M * DIN);
    float* xdb   = alloc_f((size_t)M * 64);
    _Float16* a16 = alloc_h((size_t)M * DIN);    // shared f16 activation arena
    _Float16* b16 = alloc_h((size_t)M * FF);     // ffn intermediate f16
    _Float16* w16 = alloc_h(11665408);           // f16 weight arena

    _Float16* w_tc  = w16;
    _Float16* w_in  = w_tc  + (size_t)D * 3 * D;
    _Float16* w_xp  = w_in  + (size_t)NB * 2 * DIN * D;
    _Float16* w_dt  = w_xp  + (size_t)NB * (DTR + 2 * N) * DIN;
    _Float16* w_out = w_dt  + (size_t)NB * DIN * DTR;
    _Float16* w_f1  = w_out + (size_t)NB * D * DIN;
    _Float16* w_f2  = w_f1  + (size_t)NB * FF * D;

    auto blocks = [](size_t n) { return dim3((unsigned)((n + 255) / 256)); };
    auto cvt = [&](const float* in, _Float16* out, size_t n) {
        hms_f32_to_f16<<<blocks(n), 256, 0, stream>>>(in, out, n);
    };
    auto gemm = [&](const _Float16* A, int lda, int pad_div, const _Float16* W,
                    int K, int E, const float* bias, const float* resid,
                    float* Yf, _Float16* Yh, int ldy, int act) {
        dim3 g(M / 128, E / 64);
        hms_wmma_gemm<<<g, 128, 0, stream>>>(A, lda, pad_div, W, K, bias, resid, Yf, Yh, ldy, act);
    };
    auto lnorm = [&](const float* in, const float* w, const float* b,
                     float* outf, _Float16* outh) {
        hms_layernorm<<<dim3(M), 128, 0, stream>>>(in, w, b, outf, outh, D);
    };

    // --- weight conversions to f16 (deterministic, re-done each call) ---
    hms_tconv_w_rearrange<<<blocks((size_t)D * D * 3), 256, 0, stream>>>(tconv_w, w_tc);
    cvt(in_proj_w,  w_in,  (size_t)NB * 2 * DIN * D);
    cvt(x_proj_w,   w_xp,  (size_t)NB * (DTR + 2 * N) * DIN);
    cvt(dt_proj_w,  w_dt,  (size_t)NB * DIN * DTR);
    cvt(out_proj_w, w_out, (size_t)NB * D * DIN);
    cvt(ffn_w1,     w_f1,  (size_t)NB * FF * D);
    cvt(ffn_w2,     w_f2,  (size_t)NB * D * FF);

    // --- temporal conv block: conv(x)+bias+x, then LN -> h ---
    hms_build_xpad<<<blocks((size_t)B * (L + 2) * D), 256, 0, stream>>>(x, a16);
    gemm(a16, D, L, w_tc, 3 * D, D, tconv_b, /*resid=*/x, zbuf, nullptr, D, 0);
    lnorm(zbuf, tnorm_w, tnorm_b, hbuf, nullptr);

    // --- NB Mamba+FFN blocks ---
    for (int i = 0; i < NB; ++i) {
        float* dtbuf = u_raw;    // u_raw dead after dwconv -> reuse for dt

        // ln1(h) -> f16
        lnorm(hbuf, ln1_w + (size_t)i * D, ln1_b + (size_t)i * D, nullptr, a16);
        // in_proj: u_raw = x @ W_u^T ; z = x @ W_z^T
        gemm(a16, D, 0, w_in + (size_t)i * 2 * DIN * D,                   D, DIN, nullptr, nullptr, u_raw, nullptr, DIN, 0);
        gemm(a16, D, 0, w_in + (size_t)i * 2 * DIN * D + (size_t)DIN * D, D, DIN, nullptr, nullptr, zbuf,  nullptr, DIN, 0);
        // causal depthwise conv + silu -> u (f32 + f16)
        hms_dwconv_silu<<<blocks((size_t)M * DIN), 256, 0, stream>>>(
            u_raw, conv_w + (size_t)i * DIN * DCONV, conv_b + (size_t)i * DIN, ubuf, a16);
        // x_proj: xdb = u @ W_x^T  (E=64: dtr|B|C)
        gemm(a16, DIN, 0, w_xp + (size_t)i * 64 * DIN, DIN, 64, nullptr, nullptr, xdb, nullptr, 64, 0);
        // dtr -> f16 (row stride 32)
        hms_slice_f16<<<blocks((size_t)M * DTR), 256, 0, stream>>>(xdb, 64, a16, DTR, M);
        // dt = softplus(dtr @ W_dt^T + b)
        gemm(a16, DTR, 0, w_dt + (size_t)i * DIN * DTR, DTR, DIN,
             dt_proj_b + (size_t)i * DIN, nullptr, dtbuf, nullptr, DIN, 3);
        // selective scan
        hms_scan<<<dim3(B * DIN / 256), 256, 0, stream>>>(
            ubuf, dtbuf, xdb, A_log + (size_t)i * DIN * N, D_param + (size_t)i * DIN, ybuf);
        // gate: a16 = f16( y * silu(z) )
        hms_gate<<<blocks((size_t)M * DIN), 256, 0, stream>>>(ybuf, zbuf, a16, (size_t)M * DIN);
        // out_proj + residual into h
        gemm(a16, DIN, 0, w_out + (size_t)i * D * DIN, DIN, D, nullptr, hbuf, hbuf, nullptr, D, 0);

        // FFN: hn = ln2(h); ff16 = gelu(hn @ W1^T + b1); h += ff16 @ W2^T + b2
        lnorm(hbuf, ln2_w + (size_t)i * D, ln2_b + (size_t)i * D, nullptr, a16);
        gemm(a16, D, 0, w_f1 + (size_t)i * FF * D, D, FF, ffn_b1 + (size_t)i * FF,
             nullptr, nullptr, b16, FF, 2);
        gemm(b16, FF, 0, w_f2 + (size_t)i * D * FF, FF, D, ffn_b2 + (size_t)i * D,
             hbuf, hbuf, nullptr, D, 0);
    }
}
